// CharRNN_86792699118059
// MI455X (gfx1250) — matrix operand order repaired and checked by reading, not compiled, not run on hardware
//
#include <hip/hip_runtime.h>
#include <hip/hip_bf16.h>

typedef __attribute__((ext_vector_type(16))) _Float16 v16h;
typedef __attribute__((ext_vector_type(8)))  _Float16 v8h;
typedef __attribute__((ext_vector_type(8)))  float    v8f;

#define VOCAB 32
#define EMB   32
#define HID   128
#define BATCH 512
#define SEQ   1024

#define BM     16      // batch rows per workgroup (WMMA M)
#define HROW   136     // padded f16 stride for h rows in LDS (16B aligned, bank-spread)
#define TCHUNK 64      // index prefetch chunk length (steps)
#define NTHR   320     // 8 compute waves + 2 logits waves

// CDNA5 hardware tanh (single TRANS32 op).
static __device__ __forceinline__ float hw_tanh(float v) {
#if __has_builtin(__builtin_amdgcn_tanhf)
  return __builtin_amdgcn_tanhf(v);
#elif __has_builtin(__builtin_amdgcn_tanh_f32)
  return __builtin_amdgcn_tanh_f32(v);
#else
  float r;
  asm("v_tanh_f32 %0, %1" : "=v"(r) : "v"(v));
  return r;
#endif
}

__global__ __launch_bounds__(NTHR) void charrnn_fused(
    const int*   __restrict__ x,
    const float* __restrict__ emb,
    const float* __restrict__ Wxh,
    const float* __restrict__ bh,
    const float* __restrict__ Whh,
    const float* __restrict__ Why,
    const float* __restrict__ by,
    float*       __restrict__ out)
{
  __shared__ float    table[VOCAB * HID];   // 16 KB: (embedding @ W_xh) + b_h
  __shared__ _Float16 hbuf[2][BM * HROW];   // 8.5 KB: double-buffered hidden state (f16)
  __shared__ int      idxc[BM * TCHUNK];    // 4 KB: prefetched token indices

  const int tid  = threadIdx.x;
  const int wave = tid >> 5;          // 0..7 compute, 8..9 logits
  const int lane = tid & 31;
  const int hi   = lane >> 4;         // half-wave
  const int ln   = lane & 15;
  const int b0   = blockIdx.x * BM;
  const bool isCompute = (wave < 8);
  const int ncol = wave * 16 + ln;           // compute waves: hidden column
  const int lcol = (wave - 8) * 16 + ln;     // logits waves: vocab column

  // ---- one-time: collapse input projection into a vocab lookup table ----
  for (int i = tid; i < VOCAB * HID; i += NTHR) {
    const int v = i >> 7, n = i & (HID - 1);
    float acc = bh[n];
#pragma unroll
    for (int e = 0; e < EMB; ++e) acc += emb[v * EMB + e] * Wxh[e * HID + n];
    table[i] = acc;
  }
  for (int i = tid; i < BM * HROW; i += NTHR) {
    hbuf[0][i] = (_Float16)0.0f;
    hbuf[1][i] = (_Float16)0.0f;
  }

  // ---- one-time: resident B fragments (kept in VGPRs for all 1024 steps) ----
  // B (32x16, KxN), contiguous k order used here: lane holds col N; halves j=0..15 -> K = K0 + j
  v16h bhh[4];                 // compute waves: W_hh column slab
  v16h bhy[4];                 // logits waves:  W_hy column slab
  float byv = 0.0f;
  if (isCompute) {
#pragma unroll
    for (int kk = 0; kk < 4; ++kk) {
      const int K0 = kk * 32 + hi * 16;
#pragma unroll
      for (int j = 0; j < 16; ++j)
        bhh[kk][j] = (_Float16)Whh[(K0 + j) * HID + ncol];
    }
  } else {
#pragma unroll
    for (int kk = 0; kk < 4; ++kk) {
      const int K0 = kk * 32 + hi * 16;
#pragma unroll
      for (int j = 0; j < 16; ++j)
        bhy[kk][j] = (_Float16)Why[(K0 + j) * VOCAB + lcol];
    }
    byv = by[lcol];
  }
  __syncthreads();

  const int kb = hi * 8;           // A-fragment K sub-base for this half-wave
  float xhv[8];                    // software-pipelined xh lookups for current step
  int cur = 0;
  for (int t = 0; t < SEQ; ++t) {
    if ((t & (TCHUNK - 1)) == 0) {           // refill token-index chunk
      __syncthreads();
      for (int i = tid; i < BM * TCHUNK; i += NTHR) {
        const int m = i / TCHUNK, tc = i % TCHUNK;
        idxc[i] = x[(size_t)(b0 + m) * SEQ + t + tc];
      }
      __syncthreads();
      if (isCompute) {                       // preload xh for this step
#pragma unroll
        for (int r = 0; r < 8; ++r)
          xhv[r] = table[idxc[(r + hi * 8) * TCHUNK + (t & (TCHUNK - 1))] * HID + ncol];
      }
    }
    const int nxt = cur ^ 1;

    if (isCompute) {
      // A fragments of h_{t-1} (16x32, same contiguous k order as B): lane row M=ln;
      // halves 0..15 -> K = K0+hi*16+j
      const _Float16* hrow = &hbuf[cur][ln * HROW];
      v8f d = {0.f,0.f,0.f,0.f,0.f,0.f,0.f,0.f};
#pragma unroll
      for (int kk = 0; kk < 4; ++kk) {
        const int base = kk * 32 + 2 * kb;
        v8h lo = *(const v8h*)(hrow + base);
        v8h hc = *(const v8h*)(hrow + base + 8);
        v16h a = __builtin_shufflevector(lo, hc,
                   0,1,2,3,4,5,6,7,8,9,10,11,12,13,14,15);
        d = __builtin_amdgcn_wmma_f32_16x16x32_f16(false, a, false, bhh[kk],
                                                   (short)0, d, false, false);
      }
      // h_t = tanh(xh + h_{t-1} @ W_hh)
#pragma unroll
      for (int r = 0; r < 8; ++r) {
        const int m = r + hi * 8;              // D layout: (M=r+8*hi, N=ln)
        const float ht = hw_tanh(d[r] + xhv[r]);
        d[r] = ht;
        hbuf[nxt][m * HROW + ncol] = (_Float16)ht;
      }
      if (t == SEQ - 1) {                      // final hidden state (f32)
#pragma unroll
        for (int r = 0; r < 8; ++r) {
          const int m = r + hi * 8;
          out[(size_t)BATCH * SEQ * VOCAB + (size_t)(b0 + m) * HID + ncol] = d[r];
        }
      }
    }
    __syncthreads();

    if (!isCompute) {
      // logits_t = h_t @ W_hy + b_y  (runs in the shadow of the next step)
      const _Float16* hr2 = &hbuf[nxt][ln * HROW];
      v8f dl = {0.f,0.f,0.f,0.f,0.f,0.f,0.f,0.f};
#pragma unroll
      for (int kk = 0; kk < 4; ++kk) {
        const int base = kk * 32 + 2 * kb;
        v8h lo = *(const v8h*)(hr2 + base);
        v8h hc = *(const v8h*)(hr2 + base + 8);
        v16h a = __builtin_shufflevector(lo, hc,
                   0,1,2,3,4,5,6,7,8,9,10,11,12,13,14,15);
        dl = __builtin_amdgcn_wmma_f32_16x16x32_f16(false, a, false, bhy[kk],
                                                    (short)0, dl, false, false);
      }
#pragma unroll
      for (int r = 0; r < 8; ++r) {
        const int m = r + hi * 8;
        out[(size_t)(b0 + m) * (SEQ * VOCAB) + (size_t)t * VOCAB + lcol] = dl[r] + byv;
      }
    } else if (((t + 1) & (TCHUNK - 1)) != 0) {
      // preload xh for step t+1 (indices/table are read-only LDS -> off critical path)
      const int tc = (t + 1) & (TCHUNK - 1);
#pragma unroll
      for (int r = 0; r < 8; ++r)
        xhv[r] = table[idxc[(r + hi * 8) * TCHUNK + tc] * HID + ncol];
    }
    cur = nxt;
  }
}

extern "C" void kernel_launch(void* const* d_in, const int* in_sizes, int n_in,
                              void* d_out, int out_size, void* d_ws, size_t ws_size,
                              hipStream_t stream) {
  const int*   x   = (const int*)  d_in[0];
  const float* emb = (const float*)d_in[1];
  const float* Wxh = (const float*)d_in[2];
  const float* bh  = (const float*)d_in[3];
  const float* Whh = (const float*)d_in[4];
  const float* Why = (const float*)d_in[5];
  const float* by  = (const float*)d_in[6];
  charrnn_fused<<<BATCH / BM, NTHR, 0, stream>>>(x, emb, Wxh, bh, Whh, Why, by,
                                                 (float*)d_out);
}
